// TransformerLayer_25812753449425
// MI455X (gfx1250) — matrix operand order repaired and checked by reading, not compiled, not run on hardware
//
#include <hip/hip_runtime.h>
#include <hip/hip_bf16.h>

// ---------------------------------------------------------------------------
// Types
// ---------------------------------------------------------------------------
typedef __bf16 bf16_t;
typedef __attribute__((ext_vector_type(16))) __bf16 v16bf;
typedef __attribute__((ext_vector_type(8)))  __bf16 v8bf;
typedef __attribute__((ext_vector_type(8)))  float  v8f;

static __device__ inline v8f wmma_bf16(v16bf a, v16bf b, v8f c) {
  // D = A(16x32 bf16) * B(32x16 bf16) + C(16x16 f32)
  return __builtin_amdgcn_wmma_f32_16x16x32_bf16(
      /*neg_a=*/false, a, /*neg_b=*/false, b,
      /*c_mod=*/(short)0, c, /*reuse_a=*/false, /*reuse_b=*/false);
}

static __device__ inline v16bf cat8(v8bf lo, v8bf hi) {
  return __builtin_shufflevector(lo, hi, 0,1,2,3,4,5,6,7,8,9,10,11,12,13,14,15);
}

// ---------------------------------------------------------------------------
// Model constants
// ---------------------------------------------------------------------------
#define SEQ   2048
#define BATCH 2
#define HID   2048
#define NHEAD 16
#define HD    128
#define FFN   3072
#define NTOK  (SEQ * BATCH)        // 4096 tokens, token t = s*BATCH + b
#define QKVW  (3 * HID)            // 6144, qkv row stride
#define ATT_SCALE 0.0883883476483184405f  // 1/sqrt(128)

// ---------------------------------------------------------------------------
// fp32 -> bf16 conversion (weights)
// ---------------------------------------------------------------------------
__global__ void f32_to_bf16_kernel(const float* __restrict__ in,
                                   bf16_t* __restrict__ out, int n) {
  int i = blockIdx.x * blockDim.x + threadIdx.x;
  int stride = gridDim.x * blockDim.x;
  for (; i < n; i += stride) out[i] = (bf16_t)in[i];
}

// ---------------------------------------------------------------------------
// LayerNorm over H, one block per token, bf16 output
// ---------------------------------------------------------------------------
__global__ __launch_bounds__(256) void layernorm_bf16_kernel(
    const float* __restrict__ x, const float* __restrict__ g,
    const float* __restrict__ b, bf16_t* __restrict__ out, int Hdim) {
  __shared__ float s_sum[256];
  __shared__ float s_sq[256];
  const int t = blockIdx.x;
  const float* row = x + (size_t)t * Hdim;
  float s = 0.f, sq = 0.f;
  for (int i = threadIdx.x; i < Hdim; i += 256) {
    float v = row[i]; s += v; sq += v * v;
  }
  s_sum[threadIdx.x] = s; s_sq[threadIdx.x] = sq;
  __syncthreads();
  for (int off = 128; off > 0; off >>= 1) {
    if ((int)threadIdx.x < off) {
      s_sum[threadIdx.x] += s_sum[threadIdx.x + off];
      s_sq[threadIdx.x]  += s_sq[threadIdx.x + off];
    }
    __syncthreads();
  }
  const float mean = s_sum[0] / (float)Hdim;
  const float var  = s_sq[0] / (float)Hdim - mean * mean;
  const float rstd = rsqrtf(var + 1e-5f);
  for (int i = threadIdx.x; i < Hdim; i += 256) {
    float v = (row[i] - mean) * rstd * g[i] + b[i];
    out[(size_t)t * Hdim + i] = (bf16_t)v;
  }
}

// ---------------------------------------------------------------------------
// WMMA GEMM: C[M,N] = A[M,K] * W[N,K]^T  (+ bias, + optional residual/gelu)
//   MODE 0: out_bf16 = bf16(C + bias)
//   MODE 1: out_f32  = resid + C + bias
//   MODE 2: out_bf16 = bf16(gelu_tanh(C + bias))
// Block: 256 threads = 8 waves (2 m x 4 n); wave tile 32x64
// (2x4 WMMA tiles, 8 WMMA per k-step from 6 fragments); block tile 64x256.
// M%64==0, N%256==0, K%32==0 assumed.
// Weights (bf16, ~59MB total) live in the 192MB L2 -> stream from global.
// ---------------------------------------------------------------------------
template <int MODE>
__global__ __launch_bounds__(256) void gemm_bf16_kernel(
    const bf16_t* __restrict__ A, const bf16_t* __restrict__ Bw,
    const float* __restrict__ bias, const float* __restrict__ resid,
    void* __restrict__ outp, int M, int N, int K) {
  const int lane   = threadIdx.x & 31;
  const int lane15 = lane & 15;
  const int half   = lane >> 4;
  const int kbA    = half * 16;   // A fragment K base within 32-chunk
  const int kbB    = half * 16;   // B fragment K base within 32-chunk
  const int wave   = threadIdx.x >> 5;
  const int wm     = wave >> 2;   // 0..1
  const int wn     = wave & 3;    // 0..3
  const int m_base = blockIdx.y * 64 + wm * 32;
  const int n_base = blockIdx.x * 256 + wn * 64;

  const bf16_t* arow[2];
  const bf16_t* brow[4];
#pragma unroll
  for (int mi = 0; mi < 2; ++mi)
    arow[mi] = A + (size_t)(m_base + mi * 16 + lane15) * K;
#pragma unroll
  for (int ni = 0; ni < 4; ++ni)
    brow[ni] = Bw + (size_t)(n_base + ni * 16 + lane15) * K;

  const v8f vzero = {0.f,0.f,0.f,0.f,0.f,0.f,0.f,0.f};
  v8f acc[2][4];
#pragma unroll
  for (int mi = 0; mi < 2; ++mi)
#pragma unroll
    for (int ni = 0; ni < 4; ++ni) acc[mi][ni] = vzero;

  for (int k0 = 0; k0 < K; k0 += 32) {
    v16bf af[2], bf[4];
#pragma unroll
    for (int mi = 0; mi < 2; ++mi) {
      v8bf lo = *(const v8bf*)(arow[mi] + k0 + kbA);
      v8bf hi = *(const v8bf*)(arow[mi] + k0 + 8 + kbA);
      af[mi] = cat8(lo, hi);
    }
#pragma unroll
    for (int ni = 0; ni < 4; ++ni)
      bf[ni] = *(const v16bf*)(brow[ni] + k0 + kbB);
#pragma unroll
    for (int mi = 0; mi < 2; ++mi)
#pragma unroll
      for (int ni = 0; ni < 4; ++ni)
        acc[mi][ni] = wmma_bf16(af[mi], bf[ni], acc[mi][ni]);
  }

  // Epilogue: C layout -> element (r, lane): row = half*8 + r, col = lane15
#pragma unroll
  for (int mi = 0; mi < 2; ++mi) {
#pragma unroll
    for (int ni = 0; ni < 4; ++ni) {
      const int col = n_base + ni * 16 + lane15;
      const float bcol = bias[col];
#pragma unroll
      for (int r = 0; r < 8; ++r) {
        const int row = m_base + mi * 16 + half * 8 + r;
        const size_t idx = (size_t)row * N + col;
        float v = acc[mi][ni][r] + bcol;
        if (MODE == 1) {
          ((float*)outp)[idx] = resid[idx] + v;
        } else if (MODE == 2) {
          float x = v;
          float t = 0.7978845608028654f * (x + 0.044715f * x * x * x);
          float gl = 0.5f * x * (1.0f + tanhf(t));
          ((bf16_t*)outp)[idx] = (bf16_t)gl;
        } else {
          ((bf16_t*)outp)[idx] = (bf16_t)v;
        }
      }
    }
  }
}

// ---------------------------------------------------------------------------
// Flash attention: one block per (q-tile of 64, head, batch); 4 waves,
// each wave owns 16 q-rows. Online softmax with causal mask.
// qkv layout: row t = s*BATCH + b, 6144 cols = [head][q(128) k(128) v(128)]
// ctx layout: row t, 2048 cols = [head][d(128)]
// ---------------------------------------------------------------------------
#define KROWSTEP ((size_t)16 * BATCH * QKVW)  // 16 key rows in elements

__global__ __launch_bounds__(128) void attention_kernel(
    const bf16_t* __restrict__ qkv, bf16_t* __restrict__ ctx) {
  const int qt    = blockIdx.x;    // 0..31
  const int head  = blockIdx.y;    // 0..15
  const int batch = blockIdx.z;    // 0..1
  const int lane   = threadIdx.x & 31;
  const int lane15 = lane & 15;
  const int half   = lane >> 4;
  const int wave   = threadIdx.x >> 5;
  const int kbA    = half * 16;
  const int kbB    = half * 16;

  __shared__ __align__(32) bf16_t ldsP[4][16 * 64];   // per-wave P tile
  __shared__ __align__(32) bf16_t ldsVt[HD * 64];     // V^T tile [d][key]

  const int q0 = qt * 64 + wave * 16;
  const size_t headoff = (size_t)head * 384;

  // Q fragments (A layout), 4 K-steps of 32 over HD=128, loaded once
  v16bf qf[4];
  {
    const bf16_t* qrow = qkv + ((size_t)(q0 + lane15) * BATCH + batch) * QKVW + headoff;
#pragma unroll
    for (int ks = 0; ks < 4; ++ks) {
      v8bf lo = *(const v8bf*)(qrow + ks * 32 + kbA);
      v8bf hi = *(const v8bf*)(qrow + ks * 32 + 8 + kbA);
      qf[ks] = cat8(lo, hi);
    }
  }

  const v8f vzero = {0.f,0.f,0.f,0.f,0.f,0.f,0.f,0.f};
  float m_i[8], l_i[8];
  v8f o[8];
#pragma unroll
  for (int r = 0; r < 8; ++r) { m_i[r] = -3.0e38f; l_i[r] = 0.f; }
#pragma unroll
  for (int dt = 0; dt < 8; ++dt) o[dt] = vzero;

  for (int kt = 0; kt <= qt; ++kt) {
    __syncthreads();  // all waves done reading previous V^T
    // Stage V tile transposed: ldsVt[d][key_local]
    for (int c = threadIdx.x; c < 64 * 16; c += 128) {
      const int kl  = c >> 4;          // key 0..63
      const int ds8 = (c & 15) * 8;    // d segment
      const bf16_t* vrow = qkv + ((size_t)(kt * 64 + kl) * BATCH + batch) * QKVW
                           + headoff + 256;
      v8bf vv = *(const v8bf*)(vrow + ds8);
#pragma unroll
      for (int i = 0; i < 8; ++i) ldsVt[(ds8 + i) * 64 + kl] = vv[i];
    }
    __syncthreads();

    // S = Q * K^T : ks-outer / nt-inner so the 4 fragment loads per step
    // clause together (single base + immediate offsets) and feed 4
    // independent accumulators.
    v8f sc[4];
#pragma unroll
    for (int nt = 0; nt < 4; ++nt) sc[nt] = vzero;
    {
      const bf16_t* kbase = qkv + ((size_t)(kt * 64 + lane15) * BATCH + batch) * QKVW
                            + headoff + 128 + kbB;
#pragma unroll
      for (int ks = 0; ks < 4; ++ks) {
        v16bf kf[4];
#pragma unroll
        for (int nt = 0; nt < 4; ++nt)
          kf[nt] = *(const v16bf*)(kbase + (size_t)nt * KROWSTEP + ks * 32);
#pragma unroll
        for (int nt = 0; nt < 4; ++nt)
          sc[nt] = wmma_bf16(qf[ks], kf[nt], sc[nt]);
      }
    }

    // scale + causal mask (reference: scale, then masked -> -10000)
#pragma unroll
    for (int nt = 0; nt < 4; ++nt) {
      const int colg = kt * 64 + nt * 16 + lane15;
#pragma unroll
      for (int r = 0; r < 8; ++r) {
        const int rowg = q0 + half * 8 + r;
        float v = sc[nt][r] * ATT_SCALE;
        sc[nt][r] = (colg > rowg) ? -10000.0f : v;
      }
    }

    // online softmax: row reductions across 16 lanes of the N group
    float rmax[8];
#pragma unroll
    for (int r = 0; r < 8; ++r)
      rmax[r] = fmaxf(fmaxf(sc[0][r], sc[1][r]), fmaxf(sc[2][r], sc[3][r]));
#pragma unroll
    for (int msk = 1; msk < 16; msk <<= 1)
#pragma unroll
      for (int r = 0; r < 8; ++r)
        rmax[r] = fmaxf(rmax[r], __shfl_xor(rmax[r], msk, 32));

    float alpha[8];
#pragma unroll
    for (int r = 0; r < 8; ++r) {
      float mn = fmaxf(m_i[r], rmax[r]);
      alpha[r] = __expf(m_i[r] - mn);
      m_i[r] = mn;
    }
    float rsum[8];
#pragma unroll
    for (int nt = 0; nt < 4; ++nt)
#pragma unroll
      for (int r = 0; r < 8; ++r)
        sc[nt][r] = __expf(sc[nt][r] - m_i[r]);
#pragma unroll
    for (int r = 0; r < 8; ++r)
      rsum[r] = sc[0][r] + sc[1][r] + sc[2][r] + sc[3][r];
#pragma unroll
    for (int msk = 1; msk < 16; msk <<= 1)
#pragma unroll
      for (int r = 0; r < 8; ++r)
        rsum[r] += __shfl_xor(rsum[r], msk, 32);
#pragma unroll
    for (int r = 0; r < 8; ++r) l_i[r] = l_i[r] * alpha[r] + rsum[r];
#pragma unroll
    for (int dt = 0; dt < 8; ++dt)
#pragma unroll
      for (int r = 0; r < 8; ++r) o[dt][r] *= alpha[r];

    // P (C layout, f32) -> bf16 A layout via per-wave LDS region
#pragma unroll
    for (int nt = 0; nt < 4; ++nt)
#pragma unroll
      for (int r = 0; r < 8; ++r)
        ldsP[wave][(half * 8 + r) * 64 + nt * 16 + lane15] = (bf16_t)sc[nt][r];
    // same-wave LDS RAW: DS ops are in-order; compiler inserts s_wait_dscnt

    v16bf pf[2];
#pragma unroll
    for (int kps = 0; kps < 2; ++kps) {
      v8bf lo = *(const v8bf*)&ldsP[wave][lane15 * 64 + kps * 32 + kbA];
      v8bf hi = *(const v8bf*)&ldsP[wave][lane15 * 64 + kps * 32 + 8 + kbA];
      pf[kps] = cat8(lo, hi);
    }
    // O += P * V : kps-outer / dt-inner -> 8 independent accumulation chains
#pragma unroll
    for (int kps = 0; kps < 2; ++kps) {
#pragma unroll
      for (int dt = 0; dt < 8; ++dt) {
        v16bf vf = *(const v16bf*)&ldsVt[(dt * 16 + lane15) * 64 + kps * 32 + kbB];
        o[dt] = wmma_bf16(pf[kps], vf, o[dt]);
      }
    }
  }

  // Epilogue: context[t, head*128 + d] = O / l
#pragma unroll
  for (int dt = 0; dt < 8; ++dt) {
#pragma unroll
    for (int r = 0; r < 8; ++r) {
      const int s = q0 + half * 8 + r;
      float val = o[dt][r] / l_i[r];
      ctx[((size_t)s * BATCH + batch) * HID + head * HD + dt * 16 + lane15] =
          (bf16_t)val;
    }
  }
}

// ---------------------------------------------------------------------------
// Launch
// ---------------------------------------------------------------------------
extern "C" void kernel_launch(void* const* d_in, const int* in_sizes, int n_in,
                              void* d_out, int out_size, void* d_ws, size_t ws_size,
                              hipStream_t stream) {
  (void)in_sizes; (void)n_in; (void)out_size; (void)ws_size;
  const float* hidden  = (const float*)d_in[0];
  // d_in[1] attention_mask: causal, derived analytically
  const float* w_qkv   = (const float*)d_in[2];
  const float* b_qkv   = (const float*)d_in[3];
  const float* w_dense = (const float*)d_in[4];
  const float* b_dense = (const float*)d_in[5];
  const float* w_fc1   = (const float*)d_in[6];
  const float* b_fc1   = (const float*)d_in[7];
  const float* w_fc2   = (const float*)d_in[8];
  const float* b_fc2   = (const float*)d_in[9];
  const float* ln1_g   = (const float*)d_in[10];
  const float* ln1_b   = (const float*)d_in[11];
  const float* ln2_g   = (const float*)d_in[12];
  const float* ln2_b   = (const float*)d_in[13];
  float* out = (float*)d_out;

  // Workspace carve-up (~208 MB total)
  char* ws = (char*)d_ws;
  size_t off = 0;
  auto carve = [&](size_t bytes) -> void* {
    void* p = ws + off;
    off += (bytes + 255) & ~(size_t)255;
    return p;
  };
  bf16_t* wqkv_b   = (bf16_t*)carve((size_t)QKVW * HID * 2);   // 25.2 MB
  bf16_t* wdense_b = (bf16_t*)carve((size_t)HID * HID * 2);    //  8.4 MB
  bf16_t* wfc1_b   = (bf16_t*)carve((size_t)FFN * HID * 2);    // 12.6 MB
  bf16_t* wfc2_b   = (bf16_t*)carve((size_t)HID * FFN * 2);    // 12.6 MB
  bf16_t* ln1_o    = (bf16_t*)carve((size_t)NTOK * HID * 2);   // 16.8 MB
  bf16_t* qkv_o    = (bf16_t*)carve((size_t)NTOK * QKVW * 2);  // 50.3 MB
  bf16_t* ctx_o    = (bf16_t*)carve((size_t)NTOK * HID * 2);   // 16.8 MB
  float*  x1       = (float*) carve((size_t)NTOK * HID * 4);   // 33.6 MB
  bf16_t* ln2_o    = (bf16_t*)carve((size_t)NTOK * HID * 2);   // 16.8 MB
  bf16_t* inter_o  = (bf16_t*)carve((size_t)NTOK * FFN * 2);   // 25.2 MB

  // Weight conversions to bf16
  f32_to_bf16_kernel<<<dim3(2048), 256, 0, stream>>>(w_qkv, wqkv_b, QKVW * HID);
  f32_to_bf16_kernel<<<dim3(2048), 256, 0, stream>>>(w_dense, wdense_b, HID * HID);
  f32_to_bf16_kernel<<<dim3(2048), 256, 0, stream>>>(w_fc1, wfc1_b, FFN * HID);
  f32_to_bf16_kernel<<<dim3(2048), 256, 0, stream>>>(w_fc2, wfc2_b, HID * FFN);

  // ln1
  layernorm_bf16_kernel<<<dim3(NTOK), 256, 0, stream>>>(hidden, ln1_g, ln1_b,
                                                        ln1_o, HID);
  // qkv = ln1 @ w_qkv^T + b_qkv  -> bf16
  gemm_bf16_kernel<0><<<dim3(QKVW / 256, NTOK / 64), 256, 0, stream>>>(
      ln1_o, wqkv_b, b_qkv, nullptr, qkv_o, NTOK, QKVW, HID);
  // flash attention -> context (bf16)
  attention_kernel<<<dim3(SEQ / 64, NHEAD, BATCH), 128, 0, stream>>>(qkv_o, ctx_o);
  // x1 = hidden + ctx @ w_dense^T + b_dense  -> f32
  gemm_bf16_kernel<1><<<dim3(HID / 256, NTOK / 64), 256, 0, stream>>>(
      ctx_o, wdense_b, b_dense, hidden, x1, NTOK, HID, HID);
  // ln2
  layernorm_bf16_kernel<<<dim3(NTOK), 256, 0, stream>>>(x1, ln2_g, ln2_b,
                                                        ln2_o, HID);
  // inter = gelu(ln2 @ w_fc1^T + b_fc1) -> bf16
  gemm_bf16_kernel<2><<<dim3(FFN / 256, NTOK / 64), 256, 0, stream>>>(
      ln2_o, wfc1_b, b_fc1, nullptr, inter_o, NTOK, FFN, HID);
  // out = x1 + inter @ w_fc2^T + b_fc2 -> f32
  gemm_bf16_kernel<1><<<dim3(HID / 256, NTOK / 64), 256, 0, stream>>>(
      inter_o, wfc2_b, b_fc2, x1, out, NTOK, HID, FFN);
}
